// MultiHeadCausalSelfAttention_20822001451672
// MI455X (gfx1250) — matrix operand order repaired and checked by reading, not compiled, not run on hardware
//
#include <hip/hip_runtime.h>

// ---------------------------------------------------------------------------
// Types / WMMA helpers (CDNA5 gfx1250, wave32)
// ---------------------------------------------------------------------------
typedef __attribute__((ext_vector_type(16))) __bf16 v16bf;
typedef __attribute__((ext_vector_type(8)))  __bf16 v8bf;
typedef __attribute__((ext_vector_type(8)))  float  v8f;
typedef __bf16 bf16_t;

union Frag16 { v16bf v; v8bf h[2]; };

__device__ __forceinline__ v8f wmma_bf16(v16bf a, v16bf b, v8f c) {
  // D(f32 16x16) = A(bf16 16x32) * B(bf16 32x16) + C
  return __builtin_amdgcn_wmma_f32_16x16x32_bf16(
      /*neg_a=*/false, a, /*neg_b=*/false, b,
      /*c_mod=*/(short)0, c, /*reuse_a=*/false, /*reuse_b=*/false);
}

// A fragment: 16(M) x 32(K) bf16, source row-major with leading dim `ld`.
// Lanes 0-15 hold M=lane, K={0..7,16..23}; lanes 16-31: M=lane-16, K={8..15,24..31}.
__device__ __forceinline__ v16bf load_a_frag(const bf16_t* base, int ld, int lane) {
  const int m = lane & 15, hf = lane >> 4;
  const bf16_t* p = base + (size_t)m * ld + hf * 8;
  Frag16 f;
  f.h[0] = *reinterpret_cast<const v8bf*>(p);
  f.h[1] = *reinterpret_cast<const v8bf*>(p + 16);
  return f.v;
}

// B fragment: 32(K) x 16(N) bf16, source stores Bᵀ row-major (src[n][k], ld).
// Lanes 0-15 hold N=lane, K={0..7,16..23}; lanes 16-31 hold N=lane-16, K={8..15,24..31}.
__device__ __forceinline__ v16bf load_b_frag(const bf16_t* base, int ld, int lane) {
  const int n = lane & 15, hf = lane >> 4;
  const bf16_t* p = base + (size_t)n * ld + hf * 8;
  Frag16 f;
  f.h[0] = *reinterpret_cast<const v8bf*>(p);
  f.h[1] = *reinterpret_cast<const v8bf*>(p + 16);
  return f.v;
}

// Async global->LDS 16B-per-lane copy (gfx1250, ASYNCcnt-tracked).
__device__ __forceinline__ void async_copy_b128(bf16_t* lds_dst, const bf16_t* gsrc) {
  const unsigned loff = (unsigned)(uintptr_t)lds_dst;  // LDS aperture: low 32b = offset
  asm volatile("global_load_async_to_lds_b128 %0, %1, off"
               :: "v"(loff), "v"(gsrc) : "memory");
}
__device__ __forceinline__ void wait_async_all() {
  asm volatile("s_wait_asynccnt 0" ::: "memory");
}

// ---------------------------------------------------------------------------
// Problem constants
// ---------------------------------------------------------------------------
#define BATCH   4
#define SEQ     2048
#define EMB     1024
#define NHEAD   16
#define HDIM    64
#define MROWS   (BATCH * SEQ)   // 8192

// ---------------------------------------------------------------------------
// f32 -> bf16 conversion
// ---------------------------------------------------------------------------
__global__ __launch_bounds__(256) void cvt_f32_bf16(const float* __restrict__ src,
                                                    bf16_t* __restrict__ dst, int n) {
  int i = blockIdx.x * blockDim.x + threadIdx.x;
  if (i < n) dst[i] = (bf16_t)src[i];
}

// ---------------------------------------------------------------------------
// Y[M,N] = A[M,K] * W[N,K]^T  (bf16 in, fp32 accumulate)
// One wave computes a 64x64 tile: 16 WMMAs per 16 b128 loads per K-step.
// mode 0: store bf16, head-split        dst[b,h,s,d]
// mode 1: store bf16, head-split + T    dst[b,h,d,s]   (for V)
// mode 2: store fp32 row-major          dst[m,n]       (final projection)
// ---------------------------------------------------------------------------
__global__ __launch_bounds__(256) void gemm_wmma(const bf16_t* __restrict__ A,
                                                 const bf16_t* __restrict__ W,
                                                 bf16_t* __restrict__ out_bf,
                                                 float*  __restrict__ out_f,
                                                 int Mtiles, int Ntiles, int mode) {
  const int lane = threadIdx.x & 31;
  const int wid  = (blockIdx.x * blockDim.x + threadIdx.x) >> 5;
  if (wid >= Mtiles * Ntiles) return;           // wave-uniform exit (EXEC stays full)
  const int m0 = (wid / Ntiles) * 64;
  const int n0 = (wid % Ntiles) * 64;

  const v8f vzero = {0.f, 0.f, 0.f, 0.f, 0.f, 0.f, 0.f, 0.f};
  v8f acc[4][4];
#pragma unroll
  for (int i = 0; i < 4; ++i)
#pragma unroll
    for (int j = 0; j < 4; ++j) acc[i][j] = vzero;

  for (int k0 = 0; k0 < EMB; k0 += 32) {
    v16bf bfr[4];
#pragma unroll
    for (int jn = 0; jn < 4; ++jn)
      bfr[jn] = load_b_frag(W + (size_t)(n0 + jn * 16) * EMB + k0, EMB, lane);
#pragma unroll
    for (int im = 0; im < 4; ++im) {
      v16bf a = load_a_frag(A + (size_t)(m0 + im * 16) * EMB + k0, EMB, lane);
#pragma unroll
      for (int jn = 0; jn < 4; ++jn)
        acc[im][jn] = wmma_bf16(a, bfr[jn], acc[im][jn]);
    }
  }

  // Epilogue. C/D layout: element (vgpr e, lane l): M = e + (l>=16 ? 8:0), N = l&15.
  const int nl = lane & 15, hf = lane >> 4;
#pragma unroll
  for (int im = 0; im < 4; ++im) {
#pragma unroll
    for (int jn = 0; jn < 4; ++jn) {
#pragma unroll
      for (int e = 0; e < 8; ++e) {
        const int row = m0 + im * 16 + hf * 8 + e;
        const int col = n0 + jn * 16 + nl;
        const float v = acc[im][jn][e];
        if (mode == 2) {
          out_f[(size_t)row * EMB + col] = v;
        } else {
          const int b = row >> 11, s = row & (SEQ - 1);
          const int h = col >> 6,  d = col & (HDIM - 1);
          const size_t idx = (mode == 0)
              ? ((((size_t)b * NHEAD + h) * SEQ + s) * HDIM + d)
              : ((((size_t)b * NHEAD + h) * HDIM + d) * SEQ + s);
          out_bf[idx] = (bf16_t)v;
        }
      }
    }
  }
}

// ---------------------------------------------------------------------------
// Causal flash attention, LDS-shared K/V with async double buffering.
// Block = 8 waves = one (b,h) x 128-query stripe; wave w owns 16 query rows.
// Q,K: [B,H,S,D] bf16.  Vt: [B,H,D,S] bf16.  O: [B,S,E] bf16 (merged heads).
// ---------------------------------------------------------------------------
__global__ __launch_bounds__(256) void attn_flash(const bf16_t* __restrict__ Q,
                                                  const bf16_t* __restrict__ K,
                                                  const bf16_t* __restrict__ Vt,
                                                  bf16_t* __restrict__ O) {
  __shared__ bf16_t kt[2][32 * HDIM];           // 32 keys x 64 d   (4 KB each)
  __shared__ bf16_t vt[2][HDIM * 32];           // 64 d   x 32 keys (4 KB each)
  __shared__ bf16_t plds[8][16 * 32];           // per-wave P tile  (8 KB)

  const int tid  = threadIdx.x;
  const int lane = tid & 31;
  const int w    = tid >> 5;
  const int bh   = blockIdx.x >> 4;             // 16 stripes of 128 queries per (b,h)
  const int Q0   = (blockIdx.x & 15) * 128;
  const int q0   = Q0 + w * 16;

  const bf16_t* Qb = Q  + ((size_t)bh * SEQ + q0) * HDIM;
  const bf16_t* Kb = K  + (size_t)bh * SEQ * HDIM;
  const bf16_t* Vb = Vt + (size_t)bh * HDIM * SEQ;

  const v16bf aq0 = load_a_frag(Qb,      HDIM, lane);   // d 0..31
  const v16bf aq1 = load_a_frag(Qb + 32, HDIM, lane);   // d 32..63

  const v8f vzero = {0.f, 0.f, 0.f, 0.f, 0.f, 0.f, 0.f, 0.f};
  v8f o[4] = {vzero, vzero, vzero, vzero};
  float mrow[8], lrow[8];
#pragma unroll
  for (int e = 0; e < 8; ++e) { mrow[e] = -3.0e38f; lrow[e] = 0.0f; }

  const float Cs = 0.125f * 1.44269504088896340736f;    // 1/sqrt(64) * log2(e)
  const int nl = lane & 15, hf = lane >> 4;
  const int jmaxW = (q0 + 15) >> 5;                     // this wave's causal bound
  const int jmaxB = (Q0 + 127) >> 5;                    // block-wide bound

  // Cooperative async tile fill: 256 lanes x 16B cover K-tile, same for V-tile.
  const int krow = tid >> 3, kcol = (tid & 7) * 8;      // 32 x 64 bf16
  const int vrow = tid >> 2, vcol = (tid & 3) * 8;      // 64 x 32 bf16

  // Prologue: fill buffer 0 with key tile 0.
  async_copy_b128(&kt[0][krow * HDIM + kcol], Kb + (size_t)krow * HDIM + kcol);
  async_copy_b128(&vt[0][vrow * 32 + vcol],   Vb + (size_t)vrow * SEQ + vcol);
  wait_async_all();
  __syncthreads();

  for (int j = 0; j <= jmaxB; ++j) {
    const int k0  = j * 32;
    const int buf = j & 1;

    // Prefetch tile j+1 into the other buffer (freed by last iteration's barrier).
    if (j < jmaxB) {
      const int kn = k0 + 32;
      async_copy_b128(&kt[buf ^ 1][krow * HDIM + kcol],
                      Kb + (size_t)(kn + krow) * HDIM + kcol);
      async_copy_b128(&vt[buf ^ 1][vrow * 32 + vcol],
                      Vb + (size_t)vrow * SEQ + kn + vcol);
    }

    if (j <= jmaxW) {                                   // wave-uniform causal skip
      // ---- scores S = Q Kᵀ from LDS K tile ----
      v8f sc0 = vzero, sc1 = vzero;
      {
        v16bf bk;
        bk  = load_b_frag(&kt[buf][0 * HDIM + 0],       HDIM, lane);
        sc0 = wmma_bf16(aq0, bk, sc0);
        bk  = load_b_frag(&kt[buf][0 * HDIM + 32],      HDIM, lane);
        sc0 = wmma_bf16(aq1, bk, sc0);
        bk  = load_b_frag(&kt[buf][16 * HDIM + 0],      HDIM, lane);
        sc1 = wmma_bf16(aq0, bk, sc1);
        bk  = load_b_frag(&kt[buf][16 * HDIM + 32],     HDIM, lane);
        sc1 = wmma_bf16(aq1, bk, sc1);
      }

      // ---- causal mask + online softmax (row = 16 lanes of one wave half) ----
#pragma unroll
      for (int e = 0; e < 8; ++e) {
        const int row = q0 + hf * 8 + e;
        float s0 = ((k0 + nl)      <= row) ? sc0[e] : -3.0e38f;
        float s1 = ((k0 + 16 + nl) <= row) ? sc1[e] : -3.0e38f;

        float mx = fmaxf(s0, s1);
        mx = fmaxf(mx, __shfl_xor(mx, 8));
        mx = fmaxf(mx, __shfl_xor(mx, 4));
        mx = fmaxf(mx, __shfl_xor(mx, 2));
        mx = fmaxf(mx, __shfl_xor(mx, 1));

        const float mnew = fmaxf(mrow[e], mx);
        const float scl  = exp2f((mrow[e] - mnew) * Cs);
        mrow[e] = mnew;

        const float p0 = exp2f((s0 - mnew) * Cs);
        const float p1 = exp2f((s1 - mnew) * Cs);
        float sum = p0 + p1;
        sum += __shfl_xor(sum, 8);
        sum += __shfl_xor(sum, 4);
        sum += __shfl_xor(sum, 2);
        sum += __shfl_xor(sum, 1);
        lrow[e] = lrow[e] * scl + sum;

        o[0][e] *= scl; o[1][e] *= scl; o[2][e] *= scl; o[3][e] *= scl;

        const int rl = hf * 8 + e;
        plds[w][rl * 32 + nl]      = (bf16_t)p0;
        plds[w][rl * 32 + 16 + nl] = (bf16_t)p1;
      }

      // C-layout P -> A-fragment P via per-wave LDS (same-wave RAW on LDS)
      asm volatile("s_wait_dscnt 0" ::: "memory");
      Frag16 ap;
      const bf16_t* pp = &plds[w][(lane & 15) * 32 + hf * 8];
      ap.h[0] = *reinterpret_cast<const v8bf*>(pp);
      ap.h[1] = *reinterpret_cast<const v8bf*>(pp + 16);

      // ---- O += P V from LDS V tile ----
#pragma unroll
      for (int dc = 0; dc < 4; ++dc) {
        v16bf bv = load_b_frag(&vt[buf][dc * 16 * 32], 32, lane);
        o[dc] = wmma_bf16(ap.v, bv, o[dc]);
      }
    }

    // Own async loads done + everyone finished reading buf before reuse.
    wait_async_all();
    __syncthreads();
  }

  // ---- normalize and store merged heads: O[b, s, h*64 + d] ----
  const int b = bh >> 4, h = bh & (NHEAD - 1);
#pragma unroll
  for (int e = 0; e < 8; ++e) {
    const float inv = 1.0f / lrow[e];
    const int row = q0 + hf * 8 + e;
#pragma unroll
    for (int dc = 0; dc < 4; ++dc) {
      const int d = dc * 16 + nl;
      O[((size_t)b * SEQ + row) * EMB + (size_t)h * HDIM + d] = (bf16_t)(o[dc][e] * inv);
    }
  }
}

// ---------------------------------------------------------------------------
// Launch
// ---------------------------------------------------------------------------
extern "C" void kernel_launch(void* const* d_in, const int* in_sizes, int n_in,
                              void* d_out, int out_size, void* d_ws, size_t ws_size,
                              hipStream_t stream) {
  const float* x  = (const float*)d_in[0];
  const float* Wq = (const float*)d_in[1];
  const float* Wk = (const float*)d_in[2];
  const float* Wv = (const float*)d_in[3];
  const float* Wo = (const float*)d_in[4];
  float* out = (float*)d_out;

  const size_t xElems = (size_t)MROWS * EMB;        // 8,388,608
  const size_t wElems = (size_t)EMB * EMB;          // 1,048,576
  const size_t qkvBytes = xElems * sizeof(bf16_t);  // 16 MB each

  char* ws = (char*)d_ws;
  size_t off = 0;
  bf16_t* xb  = (bf16_t*)(ws + off); off += qkvBytes;          // reused as Ob later
  bf16_t* wqb = (bf16_t*)(ws + off); off += wElems * 2;
  bf16_t* wkb = (bf16_t*)(ws + off); off += wElems * 2;
  bf16_t* wvb = (bf16_t*)(ws + off); off += wElems * 2;
  bf16_t* wob = (bf16_t*)(ws + off); off += wElems * 2;
  bf16_t* Qb  = (bf16_t*)(ws + off); off += qkvBytes;
  bf16_t* Kb  = (bf16_t*)(ws + off); off += qkvBytes;
  bf16_t* Vt  = (bf16_t*)(ws + off); off += qkvBytes;
  bf16_t* Ob  = xb;   // xb is dead after the V projection (stream-ordered)

  // 1) convert to bf16
  cvt_f32_bf16<<<(int)((xElems + 255) / 256), 256, 0, stream>>>(x,  xb,  (int)xElems);
  cvt_f32_bf16<<<(int)((wElems + 255) / 256), 256, 0, stream>>>(Wq, wqb, (int)wElems);
  cvt_f32_bf16<<<(int)((wElems + 255) / 256), 256, 0, stream>>>(Wk, wkb, (int)wElems);
  cvt_f32_bf16<<<(int)((wElems + 255) / 256), 256, 0, stream>>>(Wv, wvb, (int)wElems);
  cvt_f32_bf16<<<(int)((wElems + 255) / 256), 256, 0, stream>>>(Wo, wob, (int)wElems);

  // 2) projections: Mtiles = 8192/64 = 128, Ntiles = 1024/64 = 16 -> 2048 waves
  const int Mtiles = MROWS / 64, Ntiles = EMB / 64;
  const int gemmBlocks = (Mtiles * Ntiles) / 8;     // 8 waves / block
  gemm_wmma<<<gemmBlocks, 256, 0, stream>>>(xb, wqb, Qb, nullptr, Mtiles, Ntiles, 0);
  gemm_wmma<<<gemmBlocks, 256, 0, stream>>>(xb, wkb, Kb, nullptr, Mtiles, Ntiles, 0);
  gemm_wmma<<<gemmBlocks, 256, 0, stream>>>(xb, wvb, Vt, nullptr, Mtiles, Ntiles, 1);

  // 3) flash attention: 64 (b,h) x 16 stripes = 1024 blocks x 8 waves
  attn_flash<<<1024, 256, 0, stream>>>(Qb, Kb, Vt, Ob);

  // 4) output projection -> fp32 d_out
  gemm_wmma<<<gemmBlocks, 256, 0, stream>>>(Ob, wob, nullptr, out, Mtiles, Ntiles, 2);
}
